// CitationGAT_10230612099381
// MI455X (gfx1250) — matrix operand order repaired and checked by reading, not compiled, not run on hardware
//
#include <hip/hip_runtime.h>
#include <hip/hip_bf16.h>

typedef __bf16 bf16_t;
typedef __attribute__((ext_vector_type(16))) __bf16 v16bf;
typedef __attribute__((ext_vector_type(8)))  float  v8f;
typedef __attribute__((ext_vector_type(4)))  float  f32x4;

#define HOUT 256

// ---------------------------------------------------------------------------
// Pack 16 f32 into a v_wmma_f32_16x16x32_bf16 A fragment in this file's contiguous k order.
// (per-lane: elems 0..7 = K+0..7 of this lane's half, elems 8..15 = K+8..15)
// ---------------------------------------------------------------------------
__device__ inline v16bf cvt_a(f32x4 f0, f32x4 f1, f32x4 f2, f32x4 f3) {
  v16bf a;
  a[0]=(bf16_t)f0[0];  a[1]=(bf16_t)f0[1];  a[2]=(bf16_t)f0[2];  a[3]=(bf16_t)f0[3];
  a[4]=(bf16_t)f1[0];  a[5]=(bf16_t)f1[1];  a[6]=(bf16_t)f1[2];  a[7]=(bf16_t)f1[3];
  a[8]=(bf16_t)f2[0];  a[9]=(bf16_t)f2[1];  a[10]=(bf16_t)f2[2]; a[11]=(bf16_t)f2[3];
  a[12]=(bf16_t)f3[0]; a[13]=(bf16_t)f3[1]; a[14]=(bf16_t)f3[2]; a[15]=(bf16_t)f3[3];
  return a;
}

// ---------------------------------------------------------------------------
// GEMM: Out[M x 256] = X[M x K] (f32, cvt->bf16, streamed NT) @ Wt[256 x K] bf16
// Block: 256 threads = 8 waves. Wave tile 32x64 (2x4 WMMA frags). Block tile 64x256.
// A layout: lane<16 -> row=lane,   K 0..15 contiguous of each 32-step
//           lane>=16 -> row=lane-16, K 16..31 contiguous
// B layout (from Wt row-major): lane<16 -> col=n+lane, K k0+0..15 contiguous;
//           lane>=16 -> col=n+lane-16, K k0+16..31 contiguous (32B/lane load)
// ---------------------------------------------------------------------------
__global__ void __launch_bounds__(256)
kGemmXW(const float* __restrict__ X, const bf16_t* __restrict__ Wt,
        float* __restrict__ Out, int M, int K) {
  const int lane = threadIdx.x & 31;
  const int wid  = threadIdx.x >> 5;
  const int m0   = blockIdx.x * 64 + (wid >> 2) * 32;
  const int n0   = (wid & 3) * 64;
  const int half = lane >> 4;
  const int l15  = lane & 15;

  int ra0 = m0 + l15;      if (ra0 >= M) ra0 = M - 1;  // clamp; never stored
  int ra1 = m0 + 16 + l15; if (ra1 >= M) ra1 = M - 1;
  const float*  pa0 = X + (size_t)ra0 * K + half * 16;
  const float*  pa1 = X + (size_t)ra1 * K + half * 16;
  const bf16_t* pb0 = Wt + (size_t)(n0 + l15) * K + half * 16;
  const bf16_t* pb1 = pb0 + (size_t)16 * K;
  const bf16_t* pb2 = pb0 + (size_t)32 * K;
  const bf16_t* pb3 = pb0 + (size_t)48 * K;

  v8f acc[2][4] = {};
  for (int k0 = 0; k0 < K; k0 += 32) {
    // A: two rows per wave-half, 16 f32 each, non-temporal (X is single-use)
    f32x4 x00 = __builtin_nontemporal_load((const f32x4*)(pa0));
    f32x4 x01 = __builtin_nontemporal_load((const f32x4*)(pa0 + 4));
    f32x4 x02 = __builtin_nontemporal_load((const f32x4*)(pa0 + 8));
    f32x4 x03 = __builtin_nontemporal_load((const f32x4*)(pa0 + 12));
    f32x4 x10 = __builtin_nontemporal_load((const f32x4*)(pa1));
    f32x4 x11 = __builtin_nontemporal_load((const f32x4*)(pa1 + 4));
    f32x4 x12 = __builtin_nontemporal_load((const f32x4*)(pa1 + 8));
    f32x4 x13 = __builtin_nontemporal_load((const f32x4*)(pa1 + 12));
    v16bf a0 = cvt_a(x00, x01, x02, x03);
    v16bf a1 = cvt_a(x10, x11, x12, x13);

    v16bf b0 = *(const v16bf*)pb0;
    v16bf b1 = *(const v16bf*)pb1;
    v16bf b2 = *(const v16bf*)pb2;
    v16bf b3 = *(const v16bf*)pb3;

    acc[0][0] = __builtin_amdgcn_wmma_f32_16x16x32_bf16(false, a0, false, b0, (short)0, acc[0][0], false, false);
    acc[1][0] = __builtin_amdgcn_wmma_f32_16x16x32_bf16(false, a1, false, b0, (short)0, acc[1][0], false, false);
    acc[0][1] = __builtin_amdgcn_wmma_f32_16x16x32_bf16(false, a0, false, b1, (short)0, acc[0][1], false, false);
    acc[1][1] = __builtin_amdgcn_wmma_f32_16x16x32_bf16(false, a1, false, b1, (short)0, acc[1][1], false, false);
    acc[0][2] = __builtin_amdgcn_wmma_f32_16x16x32_bf16(false, a0, false, b2, (short)0, acc[0][2], false, false);
    acc[1][2] = __builtin_amdgcn_wmma_f32_16x16x32_bf16(false, a1, false, b2, (short)0, acc[1][2], false, false);
    acc[0][3] = __builtin_amdgcn_wmma_f32_16x16x32_bf16(false, a0, false, b3, (short)0, acc[0][3], false, false);
    acc[1][3] = __builtin_amdgcn_wmma_f32_16x16x32_bf16(false, a1, false, b3, (short)0, acc[1][3], false, false);

    pa0 += 32; pa1 += 32; pb0 += 32; pb1 += 32; pb2 += 32; pb3 += 32;
  }

  // C/D layout: col = lane&15 ; elem r -> row = r + (lane>=16 ? 8 : 0)
  const int rbase = (lane >> 4) * 8;
  if (m0 + 32 <= M) {
    // fast path: whole wave tile in range, no guards
#pragma unroll
    for (int i = 0; i < 2; ++i)
#pragma unroll
      for (int j = 0; j < 4; ++j) {
        float* o = Out + (size_t)(m0 + i * 16 + rbase) * HOUT + n0 + j * 16 + l15;
#pragma unroll
        for (int r = 0; r < 8; ++r) o[(size_t)r * HOUT] = acc[i][j][r];
      }
  } else {
#pragma unroll
    for (int i = 0; i < 2; ++i)
#pragma unroll
      for (int j = 0; j < 4; ++j) {
        const int col = n0 + j * 16 + l15;
#pragma unroll
        for (int r = 0; r < 8; ++r) {
          const int row = m0 + i * 16 + rbase + r;
          if (row < M) Out[(size_t)row * HOUT + col] = acc[i][j][r];
        }
      }
  }
}

// ---------------------------------------------------------------------------
// W[K x Hn] f32 -> Wt[Hn x K] bf16 (transpose + convert)
// ---------------------------------------------------------------------------
__global__ void __launch_bounds__(256)
kTransposeBf16(const float* __restrict__ W, bf16_t* __restrict__ Wt, int K, int Hn) {
  int idx = blockIdx.x * 256 + threadIdx.x;
  if (idx >= K * Hn) return;
  int k = idx / Hn, n = idx - k * Hn;
  Wt[(size_t)n * K + k] = (bf16_t)W[idx];
}

// ve1 = We1 @ ae1, ve2 = We2 @ ae2   (one wave per output row)
__global__ void __launch_bounds__(256)
kVecE(const float* __restrict__ We1, const float* __restrict__ ae1,
      const float* __restrict__ We2, const float* __restrict__ ae2,
      float* __restrict__ ve1, float* __restrict__ ve2, int ED, int Hn) {
  int lane = threadIdx.x & 31, wid = threadIdx.x >> 5;
  int w = blockIdx.x * 8 + wid;
  if (w >= 2 * ED) return;
  const float* Wm = (w < ED) ? We1 : We2;
  const float* av = (w < ED) ? ae1 : ae2;
  float* outp     = (w < ED) ? ve1 : ve2;
  int k = (w < ED) ? w : (w - ED);
  const float* row = Wm + (size_t)k * Hn;
  float s = 0.f;
  for (int j = lane; j < Hn; j += 32) s += row[j] * av[j];
  for (int off = 16; off; off >>= 1) s += __shfl_down(s, off);
  if (lane == 0) outp[k] = s;
}

// aE1[e] = edge_attr[e,:] . ve1 ; aE2[e] = edge_attr[e,:] . ve2   (one wave per edge)
__global__ void __launch_bounds__(256)
kEdgeDots(const float* __restrict__ EA, const float* __restrict__ ve1,
          const float* __restrict__ ve2, float* __restrict__ aE1,
          float* __restrict__ aE2, int E, int ED) {
  int lane = threadIdx.x & 31, wid = threadIdx.x >> 5;
  int e = blockIdx.x * 8 + wid;
  if (e >= E) return;
  const float* row = EA + (size_t)e * ED;
  float s1 = 0.f, s2 = 0.f;
  for (int j = lane; j < ED; j += 32) {
    float v = __builtin_nontemporal_load(row + j);  // edge_attr is single-use
    s1 += v * ve1[j];
    s2 += v * ve2[j];
  }
  for (int off = 16; off; off >>= 1) { s1 += __shfl_down(s1, off); s2 += __shfl_down(s2, off); }
  if (lane == 0) { aE1[e] = s1; aE2[e] = s2; }
}

// aS[i] = xl[i,:].a_src ; aD[i] = xl[i,:].a_dst   (one wave per node)
__global__ void __launch_bounds__(256)
kNodeAlpha(const float* __restrict__ xl, const float* __restrict__ a_s,
           const float* __restrict__ a_d, float* __restrict__ aS,
           float* __restrict__ aD, int N) {
  int lane = threadIdx.x & 31, wid = threadIdx.x >> 5;
  int node = blockIdx.x * 8 + wid;
  if (node >= N) return;
  const float* row = xl + (size_t)node * HOUT;
  float s = 0.f, d = 0.f;
#pragma unroll
  for (int j = 0; j < HOUT / 32; ++j) {
    float v = row[lane + 32 * j];
    s += v * a_s[lane + 32 * j];
    d += v * a_d[lane + 32 * j];
  }
  for (int off = 16; off; off >>= 1) { s += __shfl_down(s, off); d += __shfl_down(d, off); }
  if (lane == 0) { aS[node] = s; aD[node] = d; }
}

__global__ void __launch_bounds__(256)
kInitF(float* __restrict__ p, float v, int n) {
  int i = blockIdx.x * 256 + threadIdx.x;
  if (i < n) p[i] = v;
}

// alpha[e] = leaky_relu(aS[src]+aD[dst]+aE[e], 0.2); m[dst] = max(...)
__global__ void __launch_bounds__(256)
kEdgeLogits(const int* __restrict__ ei, const float* __restrict__ aS,
            const float* __restrict__ aD, const float* __restrict__ aE,
            float* __restrict__ alpha, float* __restrict__ m, int E) {
  int e = blockIdx.x * 256 + threadIdx.x;
  if (e >= E) return;
  int s = ei[e], d = ei[E + e];
  float a = aS[s] + aD[d] + aE[e];
  a = (a > 0.f) ? a : 0.2f * a;
  alpha[e] = a;
  if (a >= 0.f) atomicMax((int*)(m + d), __float_as_int(a));
  else          atomicMin((unsigned int*)(m + d), __float_as_uint(a));
}

// ev[e] = exp(alpha[e] - m[dst]); den[dst] += ev[e]
__global__ void __launch_bounds__(256)
kEdgeExp(const int* __restrict__ ei, const float* __restrict__ alpha,
         const float* __restrict__ m, float* __restrict__ ev,
         float* __restrict__ den, int E) {
  int e = blockIdx.x * 256 + threadIdx.x;
  if (e >= E) return;
  int d = ei[E + e];
  float v = expf(alpha[e] - m[d]);
  ev[e] = v;
  atomicAdd(den + d, v);
}

// out[dst,:] += (ev[e]/den[dst]) * xl[src,:]   (one block per edge, 256 cols)
__global__ void __launch_bounds__(256)
kAggregate(const int* __restrict__ ei, const float* __restrict__ xl,
           const float* __restrict__ ev, const float* __restrict__ den,
           float* __restrict__ out, int E) {
  int e = blockIdx.x;
  __shared__ float coef_s;
  __shared__ int s_s, d_s;
  if (threadIdx.x == 0) {
    int s = ei[e], d = ei[E + e];
    s_s = s; d_s = d;
    coef_s = ev[e] / den[d];
  }
  __syncthreads();
  int c = threadIdx.x;
  atomicAdd(out + (size_t)d_s * HOUT + c, coef_s * xl[(size_t)s_s * HOUT + c]);
}

__global__ void __launch_bounds__(256)
kBiasAct(float* __restrict__ buf, const float* __restrict__ b, int N, int relu) {
  int i = blockIdx.x * 256 + threadIdx.x;
  if (i >= N * HOUT) return;
  int c = i & (HOUT - 1);
  float v = buf[i] + b[c];
  if (relu) v = fmaxf(v, 0.f);
  buf[i] = v;
}

// out[b,:] = (titles[n0[b],:] * titles[n1[b],:]) @ Wp + bp   (block per pair)
__global__ void __launch_bounds__(256)
kHead(const float* __restrict__ titles, const int* __restrict__ nodes,
      const float* __restrict__ Wp, const float* __restrict__ bp,
      float* __restrict__ out, int B) {
  int b = blockIdx.x;
  int i0 = nodes[b], i1 = nodes[B + b];
  int t = threadIdx.x;
  float c = titles[(size_t)i0 * HOUT + t] * titles[(size_t)i1 * HOUT + t];
  __shared__ float r0[256], r1[256], r2[256];
  r0[t] = c * Wp[t * 3 + 0];
  r1[t] = c * Wp[t * 3 + 1];
  r2[t] = c * Wp[t * 3 + 2];
  __syncthreads();
  for (int off = 128; off; off >>= 1) {
    if (t < off) { r0[t] += r0[t + off]; r1[t] += r1[t + off]; r2[t] += r2[t + off]; }
    __syncthreads();
  }
  if (t == 0) {
    out[b * 3 + 0] = r0[0] + bp[0];
    out[b * 3 + 1] = r1[0] + bp[1];
    out[b * 3 + 2] = r2[0] + bp[2];
  }
}

// ---------------------------------------------------------------------------
extern "C" void kernel_launch(void* const* d_in, const int* in_sizes, int n_in,
                              void* d_out, int out_size, void* d_ws, size_t ws_size,
                              hipStream_t stream) {
  const int N = 50000, E = 200000, Fdim = 768, EDd = 768, Hd = 256, Bp = 4096;
  (void)in_sizes; (void)n_in; (void)out_size; (void)ws_size;

  const float* x  = (const float*)d_in[0];
  const float* ea = (const float*)d_in[1];
  const float* W1 = (const float*)d_in[2];
  const float* as1 = (const float*)d_in[3];
  const float* ad1 = (const float*)d_in[4];
  const float* We1 = (const float*)d_in[5];
  const float* ae1 = (const float*)d_in[6];
  const float* b1  = (const float*)d_in[7];
  const float* W2  = (const float*)d_in[8];
  const float* as2 = (const float*)d_in[9];
  const float* ad2 = (const float*)d_in[10];
  const float* We2 = (const float*)d_in[11];
  const float* ae2 = (const float*)d_in[12];
  const float* b2  = (const float*)d_in[13];
  const float* Wp  = (const float*)d_in[14];
  const float* bp  = (const float*)d_in[15];
  const int* ei    = (const int*)d_in[16];
  const int* nodes = (const int*)d_in[17];
  float* out = (float*)d_out;

  char* ws = (char*)d_ws;
  size_t off = 0;
  auto alloc = [&](size_t bytes) -> void* {
    void* p = ws + off;
    off = (off + bytes + 255) & ~(size_t)255;
    return p;
  };
  bf16_t* WT1 = (bf16_t*)alloc((size_t)Fdim * Hd * 2);
  bf16_t* WT2 = (bf16_t*)alloc((size_t)Hd * Hd * 2);
  float* VE1  = (float*)alloc((size_t)EDd * 4);
  float* VE2  = (float*)alloc((size_t)EDd * 4);
  float* AE1  = (float*)alloc((size_t)E * 4);
  float* AE2  = (float*)alloc((size_t)E * 4);
  float* ALP  = (float*)alloc((size_t)E * 4);
  float* EV   = (float*)alloc((size_t)E * 4);
  float* AS   = (float*)alloc((size_t)N * 4);
  float* AD   = (float*)alloc((size_t)N * 4);
  float* MX   = (float*)alloc((size_t)N * 4);
  float* DEN  = (float*)alloc((size_t)N * 4);
  float* bufA = (float*)alloc((size_t)N * Hd * 4);  // xl (both layers)
  float* bufB = (float*)alloc((size_t)N * Hd * 4);  // agg / h / titles

  const float NEG_INF = -__builtin_inff();
  const int gemmBlocks = (N + 63) / 64;

  // ---- prep: weights + edge-attr logits (both layers in one pass) ----
  kTransposeBf16<<<(Fdim * Hd + 255) / 256, 256, 0, stream>>>(W1, WT1, Fdim, Hd);
  kTransposeBf16<<<(Hd * Hd + 255) / 256, 256, 0, stream>>>(W2, WT2, Hd, Hd);
  kVecE<<<(2 * EDd + 7) / 8, 256, 0, stream>>>(We1, ae1, We2, ae2, VE1, VE2, EDd, Hd);
  kEdgeDots<<<(E + 7) / 8, 256, 0, stream>>>(ea, VE1, VE2, AE1, AE2, E, EDd);

  // ---- layer 1 ----
  kGemmXW<<<gemmBlocks, 256, 0, stream>>>(x, WT1, bufA, N, Fdim);
  kNodeAlpha<<<(N + 7) / 8, 256, 0, stream>>>(bufA, as1, ad1, AS, AD, N);
  kInitF<<<(N + 255) / 256, 256, 0, stream>>>(MX, NEG_INF, N);
  kInitF<<<(N + 255) / 256, 256, 0, stream>>>(DEN, 0.f, N);
  kInitF<<<(N * Hd + 255) / 256, 256, 0, stream>>>(bufB, 0.f, N * Hd);
  kEdgeLogits<<<(E + 255) / 256, 256, 0, stream>>>(ei, AS, AD, AE1, ALP, MX, E);
  kEdgeExp<<<(E + 255) / 256, 256, 0, stream>>>(ei, ALP, MX, EV, DEN, E);
  kAggregate<<<E, 256, 0, stream>>>(ei, bufA, EV, DEN, bufB, E);
  kBiasAct<<<(N * Hd + 255) / 256, 256, 0, stream>>>(bufB, b1, N, 1);  // h = relu(agg+b1)

  // ---- layer 2 ----
  kGemmXW<<<gemmBlocks, 256, 0, stream>>>(bufB, WT2, bufA, N, Hd);     // xl2
  kNodeAlpha<<<(N + 7) / 8, 256, 0, stream>>>(bufA, as2, ad2, AS, AD, N);
  kInitF<<<(N + 255) / 256, 256, 0, stream>>>(MX, NEG_INF, N);
  kInitF<<<(N + 255) / 256, 256, 0, stream>>>(DEN, 0.f, N);
  kInitF<<<(N * Hd + 255) / 256, 256, 0, stream>>>(bufB, 0.f, N * Hd); // h dead now
  kEdgeLogits<<<(E + 255) / 256, 256, 0, stream>>>(ei, AS, AD, AE2, ALP, MX, E);
  kEdgeExp<<<(E + 255) / 256, 256, 0, stream>>>(ei, ALP, MX, EV, DEN, E);
  kAggregate<<<E, 256, 0, stream>>>(ei, bufA, EV, DEN, bufB, E);
  kBiasAct<<<(N * Hd + 255) / 256, 256, 0, stream>>>(bufB, b2, N, 0);  // titles

  // ---- head ----
  kHead<<<Bp, 256, 0, stream>>>(bufB, nodes, Wp, bp, out, Bp);
}
